// GenreAgnosticInference_82918638617060
// MI455X (gfx1250) — compile-verified, matrix operand order checked by reading, not run on hardware
//
#include <hip/hip_runtime.h>

// ---------------- problem constants ----------------
#define VB    32      // batch
#define EDIM  300     // embedding dim
#define HDIM  384     // condensed hidden
#define H2    768     // 2*H (cell hidden)
#define G4DIM 3072    // 8*H gate dim
#define SEQ   16      // P == Q == 16

typedef __attribute__((ext_vector_type(16))) _Float16 v16h;
typedef __attribute__((ext_vector_type(8)))  float    v8f;

// ---------------- WMMA helpers ----------------
// A-fragment (16x32 f16, M x K) from a row-major matrix with leading dim ld.
// ISA 7.12.2: lane L holds row M=L%16; K-pack base kb=(L>=16)?8:0;
// vector elems 0..7 = K kb..kb+7, elems 8..15 = K kb+16..kb+23.
__device__ __forceinline__ v16h load_frag(const _Float16* __restrict__ M,
                                          int ld, int row0, int k0, int lane) {
  v16h f;
  const _Float16* p = M + (size_t)(row0 + (lane & 15)) * (size_t)ld
                        + (size_t)(k0 + ((lane >> 4) << 3));
#pragma unroll
  for (int t = 0; t < 8; ++t) { f[t] = p[t]; f[8 + t] = p[16 + t]; }
  return f;
}

__device__ __forceinline__ v8f wmma_f16(v16h a, v16h b, v8f c) {
  // emits v_wmma_f32_16x16x32_f16
  return __builtin_amdgcn_wmma_f32_16x16x32_f16(false, a, false, b,
                                                (short)0, c, false, false);
}

__device__ __forceinline__ float sigf(float x) { return 1.0f / (1.0f + __expf(-x)); }

// ---------------- small prep kernels ----------------
__global__ void f32_to_f16_kernel(const float* __restrict__ s,
                                  _Float16* __restrict__ d, int n) {
  int i = blockIdx.x * blockDim.x + threadIdx.x;
  if (i < n) d[i] = (_Float16)s[i];
}

__global__ void bias_sum_kernel(const float* __restrict__ a,
                                const float* __restrict__ b,
                                float* __restrict__ o, int n) {
  int i = blockIdx.x * blockDim.x + threadIdx.x;
  if (i < n) o[i] = a[i] + b[i];
}

// G[pos][b][k] = sum_e emb[tok[b][pos]][e] * W_ih[k][off + e]   (e < 300)
__global__ void gpre_kernel(const int* __restrict__ tok,
                            const float* __restrict__ emb,
                            const float* __restrict__ W_ih,
                            float* __restrict__ G, int off) {
  int idx = blockIdx.x * blockDim.x + threadIdx.x;
  if (idx >= SEQ * VB * G4DIM) return;
  int k   = idx % G4DIM;
  int rem = idx / G4DIM;
  int b   = rem % VB;
  int pos = rem / VB;
  int row = tok[b * SEQ + pos];
  const float* ev = emb + (size_t)row * EDIM;
  const float* wv = W_ih + (size_t)k * (2 * EDIM) + off;
  float acc = 0.0f;
#pragma unroll 4
  for (int e = 0; e < EDIM; ++e) acc += ev[e] * wv[e];
  G[idx] = acc;
}

// ---------------- per-diagonal GATE kernel ----------------
// grid = (ncells, 6), block = 512 (16 wave32). One quad-job (i/f/g/o tiles at
// one (mtile, n-quad)) per wave: 96 jobs per cell = 6 blocks * 16 waves.
__global__ __launch_bounds__(512) void gate_kernel(
    int d,
    const _Float16* __restrict__ Whh,    // [3072][768] f16
    const float*    __restrict__ bias,   // [3072] (b_ih + b_hh)
    const float*    __restrict__ Gp,     // [16][32][3072]
    const float*    __restrict__ Gh,     // [16][32][3072]
    const _Float16* __restrict__ Hgrid,  // [16][16][32][384] f16
    const float*    __restrict__ Cgrid,  // [16][16][32][384] f32
    _Float16*       __restrict__ Oh,     // [cells][32][768]
    _Float16*       __restrict__ Oc) {
  __shared__ _Float16 hcat[VB * H2];     // 48 KB
  const int i0 = (d > 15) ? (d - 15) : 0;
  const int i  = i0 + blockIdx.x;
  const int j  = d - i;
  const int tid = threadIdx.x;

  // stage h_cat = [H(i-1,j) | H(i,j-1)] into LDS (zeros at grid boundary)
  const _Float16* Hl = Hgrid + (size_t)(((i - 1) * SEQ + j) * VB) * HDIM;
  const _Float16* Hd = Hgrid + (size_t)((i * SEQ + (j - 1)) * VB) * HDIM;
  for (int idx = tid; idx < VB * H2; idx += 512) {
    int b = idx / H2, k = idx - b * H2;
    _Float16 v = (_Float16)0.0f;
    if (k < HDIM) { if (i > 0) v = Hl[b * HDIM + k]; }
    else          { if (j > 0) v = Hd[b * HDIM + (k - HDIM)]; }
    hcat[idx] = v;
  }
  __syncthreads();

  const int wave = tid >> 5, lane = tid & 31;
  const int slot = blockIdx.y * 16 + wave;            // 0..95
  const int mtile = slot & 1;
  const int nq    = slot >> 1;                        // 0..47

  v8f acc[4];
#pragma unroll
  for (int g = 0; g < 4; ++g) acc[g] = (v8f){0, 0, 0, 0, 0, 0, 0, 0};

  for (int ks = 0; ks < 24; ++ks) {                   // K = 768
    const int k0 = ks * 32;
    v16h a = load_frag(hcat, H2, mtile * 16, k0, lane);
#pragma unroll
    for (int g = 0; g < 4; ++g) {
      v16h b = load_frag(Whh, H2, g * H2 + nq * 16, k0, lane);
      acc[g] = wmma_f16(a, b, acc[g]);
    }
  }

  // LSTM pointwise on the quad, fully in registers
  const int nloc  = lane & 15;
  const int col   = nq * 16 + nloc;                   // 0..767
  const int mbase = mtile * 16 + ((lane >> 4) << 3);
  const size_t cellbase = (size_t)blockIdx.x * VB * H2;
  const float* Cl = Cgrid + (size_t)(((i - 1) * SEQ + j) * VB) * HDIM;
  const float* Cd = Cgrid + (size_t)((i * SEQ + (j - 1)) * VB) * HDIM;
#pragma unroll
  for (int r = 0; r < 8; ++r) {
    const int m = mbase + r;
    const size_t gp = ((size_t)(i * VB + m)) * G4DIM;
    const size_t gh = ((size_t)(j * VB + m)) * G4DIM;
    float gi = acc[0][r] + Gp[gp + col]            + Gh[gh + col]            + bias[col];
    float gf = acc[1][r] + Gp[gp + H2 + col]       + Gh[gh + H2 + col]       + bias[H2 + col];
    float gg = acc[2][r] + Gp[gp + 2 * H2 + col]   + Gh[gh + 2 * H2 + col]   + bias[2 * H2 + col];
    float go = acc[3][r] + Gp[gp + 3 * H2 + col]   + Gh[gh + 3 * H2 + col]   + bias[3 * H2 + col];
    float cprev = 0.0f;
    if (col < HDIM) { if (i > 0) cprev = Cl[(size_t)m * HDIM + col]; }
    else            { if (j > 0) cprev = Cd[(size_t)m * HDIM + (col - HDIM)]; }
    float c2 = sigf(gf) * cprev + sigf(gi) * tanhf(gg);
    float h2 = sigf(go) * tanhf(c2);
    Oh[cellbase + (size_t)m * H2 + col] = (_Float16)h2;
    Oc[cellbase + (size_t)m * H2 + col] = (_Float16)c2;
  }
}

// ---------------- per-diagonal CONDENSE kernel ----------------
// grid = (ncells, 6): y in 0..2 -> nh (oh @ Wch^T + bch), y in 3..5 -> nc.
__global__ __launch_bounds__(512) void cond_kernel(
    int d,
    const _Float16* __restrict__ Oh, const _Float16* __restrict__ Oc,
    const _Float16* __restrict__ Wch, const _Float16* __restrict__ Wcc, // [384][768] f16
    const float* __restrict__ bch, const float* __restrict__ bcc,
    _Float16* __restrict__ Hgrid, float* __restrict__ Cgrid) {
  __shared__ _Float16 src[VB * H2];      // 48 KB
  const int i0 = (d > 15) ? (d - 15) : 0;
  const int i  = i0 + blockIdx.x;
  const int j  = d - i;
  const int tid = threadIdx.x;
  const bool isC = (blockIdx.y >= 3);

  const _Float16* S = (isC ? Oc : Oh) + (size_t)blockIdx.x * VB * H2;
  for (int idx = tid; idx < VB * H2; idx += 512) src[idx] = S[idx];
  __syncthreads();

  const int wave = tid >> 5, lane = tid & 31;
  const int slot = (blockIdx.y - (isC ? 3 : 0)) * 16 + wave;  // 0..47
  const int mtile = slot & 1;
  const int ntile = slot >> 1;                                // 0..23

  const _Float16* W = isC ? Wcc : Wch;
  v8f acc = (v8f){0, 0, 0, 0, 0, 0, 0, 0};
  for (int ks = 0; ks < 24; ++ks) {
    const int k0 = ks * 32;
    v16h a = load_frag(src, H2, mtile * 16, k0, lane);
    v16h b = load_frag(W,   H2, ntile * 16, k0, lane);
    acc = wmma_f16(a, b, acc);
  }

  const int n     = ntile * 16 + (lane & 15);
  const int mbase = mtile * 16 + ((lane >> 4) << 3);
  const size_t base = ((size_t)(i * SEQ + j)) * VB * HDIM;
  const float bv = isC ? bcc[n] : bch[n];
#pragma unroll
  for (int r = 0; r < 8; ++r) {
    const int m = mbase + r;
    const float v = acc[r] + bv;
    if (isC) Cgrid[base + (size_t)m * HDIM + n] = v;
    else     Hgrid[base + (size_t)m * HDIM + n] = (_Float16)v;
  }
}

// ---------------- final MLP + softmax ----------------
__global__ __launch_bounds__(384) void mlp_kernel(
    const _Float16* __restrict__ Hgrid,
    const float* __restrict__ W1, const float* __restrict__ b1,
    const float* __restrict__ W2, const float* __restrict__ b2,
    const float* __restrict__ W3, const float* __restrict__ b3,
    float* __restrict__ out) {
  __shared__ float buf[VB * HDIM];       // 48 KB
  __shared__ float logits[VB * 3];
  const int t = threadIdx.x;             // 0..383
  const _Float16* hsrc = Hgrid + (size_t)(15 * SEQ + 15) * VB * HDIM;
  for (int idx = t; idx < VB * HDIM; idx += 384) buf[idx] = (float)hsrc[idx];
  __syncthreads();

  float reg[VB];
  // layer 1: thread t owns output column t for all batches
  for (int bb = 0; bb < VB; ++bb) {
    float acc = b1[t];
    const float* w = W1 + (size_t)t * HDIM;
    for (int k = 0; k < HDIM; ++k) acc += buf[bb * HDIM + k] * w[k];
    reg[bb] = acc > 0.0f ? acc : 0.0f;
  }
  __syncthreads();
  for (int bb = 0; bb < VB; ++bb) buf[bb * HDIM + t] = reg[bb];
  __syncthreads();
  // layer 2
  for (int bb = 0; bb < VB; ++bb) {
    float acc = b2[t];
    const float* w = W2 + (size_t)t * HDIM;
    for (int k = 0; k < HDIM; ++k) acc += buf[bb * HDIM + k] * w[k];
    reg[bb] = acc > 0.0f ? acc : 0.0f;
  }
  __syncthreads();
  for (int bb = 0; bb < VB; ++bb) buf[bb * HDIM + t] = reg[bb];
  __syncthreads();
  // layer 3 logits
  if (t < VB * 3) {
    int bb = t / 3, c = t - bb * 3;
    float acc = b3[c];
    const float* w = W3 + (size_t)c * HDIM;
    for (int k = 0; k < HDIM; ++k) acc += buf[bb * HDIM + k] * w[k];
    logits[t] = acc;
  }
  __syncthreads();
  if (t < VB) {
    float l0 = logits[t * 3], l1 = logits[t * 3 + 1], l2 = logits[t * 3 + 2];
    float mx = fmaxf(l0, fmaxf(l1, l2));
    float e0 = __expf(l0 - mx), e1 = __expf(l1 - mx), e2 = __expf(l2 - mx);
    float inv = 1.0f / (e0 + e1 + e2);
    out[t * 3 + 0] = e0 * inv;
    out[t * 3 + 1] = e1 * inv;
    out[t * 3 + 2] = e2 * inv;
  }
}

// ---------------- host orchestration ----------------
extern "C" void kernel_launch(void* const* d_in, const int* in_sizes, int n_in,
                              void* d_out, int out_size, void* d_ws, size_t ws_size,
                              hipStream_t stream) {
  const int*   premise    = (const int*)d_in[0];
  const int*   hypothesis = (const int*)d_in[1];
  const float* embedding  = (const float*)d_in[2];
  const float* W_ih       = (const float*)d_in[3];
  const float* W_hh       = (const float*)d_in[4];
  const float* b_ih       = (const float*)d_in[5];
  const float* b_hh       = (const float*)d_in[6];
  const float* Wch        = (const float*)d_in[7];
  const float* bch        = (const float*)d_in[8];
  const float* Wcc        = (const float*)d_in[9];
  const float* bcc        = (const float*)d_in[10];
  const float* W1         = (const float*)d_in[11];
  const float* b1         = (const float*)d_in[12];
  const float* W2         = (const float*)d_in[13];
  const float* b2         = (const float*)d_in[14];
  const float* W3         = (const float*)d_in[15];
  const float* b3         = (const float*)d_in[16];

  // workspace carve-out (all chunk sizes are multiples of 256 bytes)
  char* ws = (char*)d_ws;
  size_t off = 0;
  auto take = [&](size_t bytes) { char* p = ws + off; off += bytes; return p; };
  _Float16* Whh_h  = (_Float16*)take((size_t)G4DIM * H2 * 2);           // 4.72 MB
  _Float16* Wch_h  = (_Float16*)take((size_t)HDIM * H2 * 2);            // 0.59 MB
  _Float16* Wcc_h  = (_Float16*)take((size_t)HDIM * H2 * 2);            // 0.59 MB
  float*    bias   = (float*)take((size_t)G4DIM * 4);
  float*    Gp     = (float*)take((size_t)SEQ * VB * G4DIM * 4);        // 6.29 MB
  float*    Gh     = (float*)take((size_t)SEQ * VB * G4DIM * 4);        // 6.29 MB
  _Float16* Hgrid  = (_Float16*)take((size_t)SEQ * SEQ * VB * HDIM * 2);// 6.29 MB
  float*    Cgrid  = (float*)take((size_t)SEQ * SEQ * VB * HDIM * 4);   // 12.6 MB
  _Float16* Oh     = (_Float16*)take((size_t)SEQ * VB * H2 * 2);        // 0.79 MB
  _Float16* Oc     = (_Float16*)take((size_t)SEQ * VB * H2 * 2);        // 0.79 MB
  (void)ws_size; (void)in_sizes; (void)n_in; (void)out_size;

  // weight conversion + bias fold
  {
    int n = G4DIM * H2;
    f32_to_f16_kernel<<<(n + 255) / 256, 256, 0, stream>>>(W_hh, Whh_h, n);
    n = HDIM * H2;
    f32_to_f16_kernel<<<(n + 255) / 256, 256, 0, stream>>>(Wch, Wch_h, n);
    f32_to_f16_kernel<<<(n + 255) / 256, 256, 0, stream>>>(Wcc, Wcc_h, n);
    bias_sum_kernel<<<(G4DIM + 255) / 256, 256, 0, stream>>>(b_ih, b_hh, bias, G4DIM);
  }

  // recurrence-independent gate contributions from embeddings
  {
    int n = SEQ * VB * G4DIM;
    gpre_kernel<<<(n + 255) / 256, 256, 0, stream>>>(premise, embedding, W_ih, Gp, 0);
    gpre_kernel<<<(n + 255) / 256, 256, 0, stream>>>(hypothesis, embedding, W_ih, Gh, EDIM);
  }

  // anti-diagonal wavefront over the 16x16 grid
  for (int d = 0; d <= 2 * (SEQ - 1); ++d) {
    int ilo = (d > SEQ - 1) ? d - (SEQ - 1) : 0;
    int ihi = (d < SEQ - 1) ? d : SEQ - 1;
    int nc  = ihi - ilo + 1;
    gate_kernel<<<dim3(nc, 6), 512, 0, stream>>>(d, Whh_h, bias, Gp, Gh,
                                                 Hgrid, Cgrid, Oh, Oc);
    cond_kernel<<<dim3(nc, 6), 512, 0, stream>>>(d, Oh, Oc, Wch_h, Wcc_h,
                                                 bch, bcc, Hgrid, Cgrid);
  }

  // final MLP + softmax
  mlp_kernel<<<1, 384, 0, stream>>>(Hgrid, W1, b1, W2, b2, W3, b3, (float*)d_out);
}
